// S6HistoryCompressor_45165876085140
// MI455X (gfx1250) — compile-verified, matrix operand order checked by reading, not run on hardware
//
#include <hip/hip_runtime.h>
#include <hip/hip_bf16.h>

typedef __attribute__((ext_vector_type(16))) __bf16 v16bf;
typedef __attribute__((ext_vector_type(8)))  __bf16 v8bf;
typedef __attribute__((ext_vector_type(8)))  float  v8f;

// Problem sizes (fixed by the reference)
#define S_LEN   2048
#define BATCH_N 16
#define D_IN    1024
#define H_DIM   2048
#define M1      (S_LEN * BATCH_N)     // 32768 rows for the input projections
#define COMP_LEN 1024                 // max(1, int(2048*0.5))
#define STEP     2                    // 2048 / 1024
#define M2      (COMP_LEN * BATCH_N)  // 16384 rows for the output GEMM

// Scan chunking
#define CHUNKS  8
#define CLEN    (S_LEN / CHUNKS)      // 256

// GEMM tiling
constexpr int BM = 128, BN = 128, BK = 32;
constexpr int LDSS = 40;   // LDS row stride in bf16 elems (80B: 16B-aligned, conflict-free)

// ---------------------------------------------------------------------------
// CDNA5 async global->LDS copy (ASYNCcnt-tracked DMA, no VGPR round trip).
// VDST = LDS byte address (low 32 bits of generic pointer), VADDR = 64-bit global.
__device__ __forceinline__ void async_copy16(const __bf16* g, __bf16* l) {
  uint32_t lds = (uint32_t)(uintptr_t)l;
  uint64_t ga  = (uint64_t)(uintptr_t)g;
  asm volatile("global_load_async_to_lds_b128 %0, %1, off"
               :: "v"(lds), "v"(ga) : "memory");
}

// ---------------------------------------------------------------------------
// f32 -> bf16 conversion (grid-stride)
__global__ void cvt_f32_bf16(const float* __restrict__ in, __bf16* __restrict__ out, size_t n) {
  size_t i = (size_t)blockIdx.x * blockDim.x + threadIdx.x;
  size_t stride = (size_t)gridDim.x * blockDim.x;
  for (; i < n; i += stride) out[i] = (__bf16)in[i];
}

// Weight prep: wd = bf16(W_delta), wc = bf16(W_C), wu = bf16(W_in + W_B), wcmp = bf16(W_comp)
__global__ void prep_weights(const float* __restrict__ Wb, const float* __restrict__ Wc,
                             const float* __restrict__ Wd, const float* __restrict__ Win,
                             const float* __restrict__ Wcomp,
                             __bf16* __restrict__ wu, __bf16* __restrict__ wc,
                             __bf16* __restrict__ wd, __bf16* __restrict__ wcmp, size_t n) {
  size_t i = (size_t)blockIdx.x * blockDim.x + threadIdx.x;
  size_t stride = (size_t)gridDim.x * blockDim.x;
  for (; i < n; i += stride) {
    wd[i]   = (__bf16)Wd[i];
    wc[i]   = (__bf16)Wc[i];
    wu[i]   = (__bf16)(Win[i] + Wb[i]);
    wcmp[i] = (__bf16)Wcomp[i];
  }
}

// ---------------------------------------------------------------------------
// C[m,n] = sum_k A[m,k] * B[n,k]   (A: MxK row-major bf16, B: NxK row-major bf16)
// MODE: 0 = plain, 1 = +bias[n], 2 = softplus(acc + bias[n])
// Block = 128x128, 8 waves of 64x32, WMMA 16x16x32 bf16 -> f32,
// double-buffered LDS fed by async global->LDS DMA.
template <int MODE>
__global__ __launch_bounds__(256) void gemm_tn_bf16(
    const __bf16* __restrict__ A, const __bf16* __restrict__ B,
    const float* __restrict__ bias, float* __restrict__ C,
    int M, int N, int K) {
  __shared__ __align__(16) __bf16 sA[2][BM * LDSS];
  __shared__ __align__(16) __bf16 sB[2][BN * LDSS];

  const int tid  = threadIdx.x;
  const int lane = tid & 31;
  const int wave = tid >> 5;
  const int wm   = (wave >> 2) * 64;   // wave M offset within block: 0 / 64
  const int wn   = (wave & 3) * 32;    // wave N offset within block: 0/32/64/96
  const int m0   = blockIdx.y * BM;
  const int n0   = blockIdx.x * BN;

  const int r    = lane & 15;          // row within 16x16 fragment
  const int hseg = lane >> 4;          // lane-half selects K {0..7,16..23} vs {8..15,24..31}

  // Each thread stages 2 A-chunks + 2 B-chunks (16B each) per tile: 4 async ops/wave-issue.
  const int c0 = tid, c1 = tid + 256;
  const int rA0 = c0 >> 2, sA0 = (c0 & 3) * 8;
  const int rA1 = c1 >> 2, sA1 = (c1 & 3) * 8;

  const v8f vzero = {0.f, 0.f, 0.f, 0.f, 0.f, 0.f, 0.f, 0.f};
  v8f acc[4][2];
#pragma unroll
  for (int mi = 0; mi < 4; ++mi)
#pragma unroll
    for (int ni = 0; ni < 2; ++ni) acc[mi][ni] = vzero;

  // Prologue: stage tile 0 into buffer 0
  {
    const int kt = 0;
    async_copy16(A + (size_t)(m0 + rA0) * K + kt + sA0, &sA[0][rA0 * LDSS + sA0]);
    async_copy16(B + (size_t)(n0 + rA0) * K + kt + sA0, &sB[0][rA0 * LDSS + sA0]);
    async_copy16(A + (size_t)(m0 + rA1) * K + kt + sA1, &sA[0][rA1 * LDSS + sA1]);
    async_copy16(B + (size_t)(n0 + rA1) * K + kt + sA1, &sB[0][rA1 * LDSS + sA1]);
  }

  int p = 0;
  for (int kt = 0; kt < K; kt += BK) {
    const bool has_next = (kt + BK) < K;
    if (has_next) {
      const int kn = kt + BK;
      const int q = p ^ 1;
      async_copy16(A + (size_t)(m0 + rA0) * K + kn + sA0, &sA[q][rA0 * LDSS + sA0]);
      async_copy16(B + (size_t)(n0 + rA0) * K + kn + sA0, &sB[q][rA0 * LDSS + sA0]);
      async_copy16(A + (size_t)(m0 + rA1) * K + kn + sA1, &sA[q][rA1 * LDSS + sA1]);
      async_copy16(B + (size_t)(n0 + rA1) * K + kn + sA1, &sB[q][rA1 * LDSS + sA1]);
      // 8 outstanding; first 4 (current tile) complete in order -> wait <= 4
      asm volatile("s_wait_asynccnt 0x4" ::: "memory");
    } else {
      asm volatile("s_wait_asynccnt 0x0" ::: "memory");
    }
    __syncthreads();

    // Fragments: per ISA 16-bit A layout, lane half 0 holds K{0..7,16..23}, half 1 K{8..15,24..31}
    v16bf fa[4], fb[2];
#pragma unroll
    for (int mi = 0; mi < 4; ++mi) {
      const __bf16* ptr = &sA[p][(wm + mi * 16 + r) * LDSS] + hseg * 8;
      v8bf lo = *(const v8bf*)ptr;
      v8bf hi = *(const v8bf*)(ptr + 16);
      fa[mi] = __builtin_shufflevector(lo, hi, 0, 1, 2, 3, 4, 5, 6, 7,
                                       8, 9, 10, 11, 12, 13, 14, 15);
    }
#pragma unroll
    for (int ni = 0; ni < 2; ++ni) {
      const __bf16* ptr = &sB[p][(wn + ni * 16 + r) * LDSS] + hseg * 8;
      v8bf lo = *(const v8bf*)ptr;
      v8bf hi = *(const v8bf*)(ptr + 16);
      fb[ni] = __builtin_shufflevector(lo, hi, 0, 1, 2, 3, 4, 5, 6, 7,
                                       8, 9, 10, 11, 12, 13, 14, 15);
    }

#pragma unroll
    for (int mi = 0; mi < 4; ++mi)
#pragma unroll
      for (int ni = 0; ni < 2; ++ni)
        acc[mi][ni] = __builtin_amdgcn_wmma_f32_16x16x32_bf16(
            false, fa[mi], false, fb[ni], (short)0, acc[mi][ni], false, false);

    __syncthreads();   // all waves done reading buf p before it is restaged
    p ^= 1;
  }

  // Epilogue: C/D layout — lane<16: M=v, N=lane; lane>=16: M=8+v, N=lane-16
#pragma unroll
  for (int mi = 0; mi < 4; ++mi) {
#pragma unroll
    for (int ni = 0; ni < 2; ++ni) {
      const int col  = n0 + wn + ni * 16 + r;
      const int rowb = m0 + wm + mi * 16 + hseg * 8;
      const float bv = (MODE >= 1) ? bias[col] : 0.f;
#pragma unroll
      for (int v = 0; v < 8; ++v) {
        float val = acc[mi][ni][v] + bv;
        if (MODE == 2) val = (val > 20.f) ? val : __logf(1.f + __expf(val));
        C[(size_t)(rowb + v) * N + col] = val;
      }
    }
  }
}

// ---------------------------------------------------------------------------
// Chunked parallel scan over h_t = (1-d_t) h + d_t u_t  (associative: h = a*h + b)
// Phase 1: per-chunk composition (P = prod a, Q = affine offset)
__global__ void scan_phase1(const float* __restrict__ dl, const float* __restrict__ uu,
                            float* __restrict__ P, float* __restrict__ Q) {
  const int BH = BATCH_N * H_DIM;
  const int bh = blockIdx.x * blockDim.x + threadIdx.x;
  const int c  = blockIdx.y;
  float Pv = 1.f, Qv = 0.f;
  for (int s = c * CLEN; s < (c + 1) * CLEN; ++s) {
    const size_t idx = (size_t)s * BH + bh;
    const float d = dl[idx];
    const float a = 1.f - d;
    Qv = a * Qv + d * uu[idx];
    Pv *= a;
  }
  P[(size_t)c * BH + bh] = Pv;
  Q[(size_t)c * BH + bh] = Qv;
}

// Phase 2: tiny sequential combine over CHUNKS to get each chunk's starting h
__global__ void scan_combine(const float* __restrict__ P, const float* __restrict__ Q,
                             float* __restrict__ hstart) {
  const int BH = BATCH_N * H_DIM;
  const int bh = blockIdx.x * blockDim.x + threadIdx.x;
  float h = 0.f;
  for (int c = 0; c < CHUNKS; ++c) {
    hstart[(size_t)c * BH + bh] = h;
    h = P[(size_t)c * BH + bh] * h + Q[(size_t)c * BH + bh];
  }
}

// Phase 3: replay each chunk with the correct starting h, emit y = C*h at odd steps (bf16)
__global__ void scan_phase3(const float* __restrict__ dl, const float* __restrict__ uu,
                            const float* __restrict__ ca, const float* __restrict__ hstart,
                            __bf16* __restrict__ ysel) {
  const int BH = BATCH_N * H_DIM;
  const int bh = blockIdx.x * blockDim.x + threadIdx.x;
  const int c  = blockIdx.y;
  float h = hstart[(size_t)c * BH + bh];
  for (int s = c * CLEN; s < (c + 1) * CLEN; ++s) {
    const size_t idx = (size_t)s * BH + bh;
    const float d = dl[idx];
    h = (1.f - d) * h + d * uu[idx];
    if ((s % STEP) == STEP - 1) {
      ysel[(size_t)(s / STEP) * BH + bh] = (__bf16)(ca[idx] * h);
    }
  }
}

// ---------------------------------------------------------------------------
extern "C" void kernel_launch(void* const* d_in, const int* in_sizes, int n_in,
                              void* d_out, int out_size, void* d_ws, size_t ws_size,
                              hipStream_t stream) {
  (void)in_sizes; (void)n_in; (void)out_size; (void)ws_size;
  const float* x      = (const float*)d_in[0];
  // d_in[1] = A (unused by the reference forward)
  const float* W_B    = (const float*)d_in[2];
  const float* W_C    = (const float*)d_in[3];
  const float* W_d    = (const float*)d_in[4];
  const float* b_d    = (const float*)d_in[5];
  const float* W_in   = (const float*)d_in[6];
  const float* b_in   = (const float*)d_in[7];
  const float* W_comp = (const float*)d_in[8];
  const float* b_comp = (const float*)d_in[9];
  float* out = (float*)d_out;

  char* ws = (char*)d_ws;
  size_t off = 0;
  auto alloc = [&](size_t bytes) -> void* {
    void* p = ws + off;
    off = (off + bytes + 255) & ~(size_t)255;
    return p;
  };
  __bf16* xb   = (__bf16*)alloc((size_t)M1 * D_IN * 2);
  __bf16* wu   = (__bf16*)alloc((size_t)H_DIM * D_IN * 2);
  __bf16* wc   = (__bf16*)alloc((size_t)H_DIM * D_IN * 2);
  __bf16* wd   = (__bf16*)alloc((size_t)H_DIM * D_IN * 2);
  __bf16* wcmp = (__bf16*)alloc((size_t)D_IN * H_DIM * 2);
  float*  dl   = (float*)alloc((size_t)M1 * H_DIM * 4);
  float*  uu   = (float*)alloc((size_t)M1 * H_DIM * 4);
  float*  ca   = (float*)alloc((size_t)M1 * H_DIM * 4);
  __bf16* ysel = (__bf16*)alloc((size_t)M2 * H_DIM * 2);
  const int BH = BATCH_N * H_DIM;
  float*  Pc   = (float*)alloc((size_t)CHUNKS * BH * 4);
  float*  Qc   = (float*)alloc((size_t)CHUNKS * BH * 4);
  float*  hst  = (float*)alloc((size_t)CHUNKS * BH * 4);

  cvt_f32_bf16<<<1024, 256, 0, stream>>>(x, xb, (size_t)M1 * D_IN);
  prep_weights<<<1024, 256, 0, stream>>>(W_B, W_C, W_d, W_in, W_comp,
                                         wu, wc, wd, wcmp, (size_t)H_DIM * D_IN);

  // Input projections: M=32768, N=2048, K=1024
  dim3 g1(H_DIM / BN, M1 / BM);
  gemm_tn_bf16<2><<<g1, 256, 0, stream>>>(xb, wd, b_d,  dl, M1, H_DIM, D_IN);  // softplus(delta)
  gemm_tn_bf16<1><<<g1, 256, 0, stream>>>(xb, wu, b_in, uu, M1, H_DIM, D_IN);  // u = xWin + b + xWB
  gemm_tn_bf16<0><<<g1, 256, 0, stream>>>(xb, wc, nullptr, ca, M1, H_DIM, D_IN); // C_all

  // Chunked scan: 8x parallelism on the heavy passes
  dim3 gs(BH / 256, CHUNKS);
  scan_phase1<<<gs, 256, 0, stream>>>(dl, uu, Pc, Qc);
  scan_combine<<<BH / 256, 256, 0, stream>>>(Pc, Qc, hst);
  scan_phase3<<<gs, 256, 0, stream>>>(dl, uu, ca, hst, ysel);

  // Output compression: M=16384, N=1024, K=2048
  dim3 g2(D_IN / BN, M2 / BM);
  gemm_tn_bf16<1><<<g2, 256, 0, stream>>>(ysel, wcmp, b_comp, out, M2, D_IN, H_DIM);
}
